// Discriminator_lstm_17609365913783
// MI455X (gfx1250) — matrix operand order repaired and checked by reading, not compiled, not run on hardware
//
#include <hip/hip_runtime.h>
#include <hip/hip_bf16.h>
#include <math.h>

typedef __attribute__((ext_vector_type(16))) _Float16 v16h;
typedef __attribute__((ext_vector_type(8)))  _Float16 v8h;
typedef __attribute__((ext_vector_type(8)))  float    v8f;

#define BSZ   4096
#define TLEN  32
#define VOCAB 10000
#define EDIM  300
#define HDIM  300
#define G4    1200      // 4*H
#define KPAD  320       // E and H padded to multiple of 32
#define NT    75        // 1200 / 16 N-tiles
#define KT    10        // 320 / 32 K-tiles per operand half
#define TILE_HALFS 512  // 32 lanes * 16 halves per B tile

// ---------------------------------------------------------------------------
// A-operand loader, matching CDNA5 16-bit A 16x32 layout:
// lane L: m = L%16 ; halves j=0..7 -> k = 8*(L/16)+j ; j=8..15 -> k += 16
// rows are zero-padded to KPAD halves -> always in-bounds, 16B-aligned chunks
// ---------------------------------------------------------------------------
__device__ __forceinline__ v16h load_a_f16pad(const _Float16* row, int kb, int c0) {
    v8h lo = *(const v8h*)(row + kb + c0);
    v8h hi = *(const v8h*)(row + kb + c0 + 16);
    v16h a;
#pragma unroll
    for (int i = 0; i < 8; ++i) { a[i] = lo[i]; a[i + 8] = hi[i]; }
    return a;
}

// ---------------------------------------------------------------------------
// Zero initial h (padded f16) and c
// ---------------------------------------------------------------------------
__global__ void init_state(_Float16* hF16, float* cbuf) {
    int idx = blockIdx.x * blockDim.x + threadIdx.x;
    if (idx < BSZ * KPAD) hF16[idx] = (_Float16)0.f;
    if (idx < BSZ * HDIM) cbuf[idx] = 0.f;
}

// ---------------------------------------------------------------------------
// Pre-convert embedding table to padded f16: embF16[VOCAB][KPAD] (6.4 MB, L2)
// ---------------------------------------------------------------------------
__global__ void cvt_embed(const float* embed_w, _Float16* embF16) {
    int idx = blockIdx.x * blockDim.x + threadIdx.x;
    if (idx >= VOCAB * KPAD) return;
    int row = idx / KPAD;
    int k   = idx - row * KPAD;
    float v = (k < EDIM) ? embed_w[row * EDIM + k] : 0.f;
    embF16[idx] = (_Float16)v;
}

// ---------------------------------------------------------------------------
// biasSum = b_ih + b_hh ; weight-normed classifier W = g * v / ||v||_row
// ---------------------------------------------------------------------------
__global__ void prep_cls_bias(const float* b_ih, const float* b_hh,
                              const float* cls_v, const float* cls_g,
                              float* biasSum, float* Wn) {
    __shared__ float s_scale[2];
    int tid = threadIdx.x;
    if (tid < 2) {
        float s = 0.f;
        for (int k = 0; k < HDIM; ++k) { float v = cls_v[tid * HDIM + k]; s += v * v; }
        s_scale[tid] = cls_g[tid] * rsqrtf(s);
    }
    __syncthreads();
    for (int i = tid; i < G4; i += blockDim.x) biasSum[i] = b_ih[i] + b_hh[i];
    for (int i = tid; i < 2 * HDIM; i += blockDim.x) {
        int cls = i / HDIM;
        Wn[i] = cls_v[i] * s_scale[cls];
    }
}

// ---------------------------------------------------------------------------
// Pack W_ih (kt 0..9) and W_hh (kt 10..19) into WMMA B-fragment tiles.
// B 16-bit 32x16 layout: lane L: n = L%16 ; half j: k = 8*(L/16) + (j&7) + 16*(j>>3)
// Tile (kt, nt) stored contiguously: [tile][lane][j]
// ---------------------------------------------------------------------------
__global__ void pack_weights(const float* W_ih, const float* W_hh, _Float16* packedW) {
    int idx = blockIdx.x * blockDim.x + threadIdx.x;
    const int total = 2 * KT * NT * TILE_HALFS;
    if (idx >= total) return;
    int tile   = idx >> 9;        // /512
    int within = idx & 511;
    int lane = within >> 4;
    int j    = within & 15;
    int kt = tile / NT;           // 0..19
    int nt = tile % NT;
    int n = nt * 16 + (lane & 15);
    int k = ((kt % KT) * 32) + ((lane >> 4) * 8) + (j & 7) + ((j >> 3) * 16);
    const float* W = (kt < KT) ? W_ih : W_hh;     // both [1200, 300] row-major
    float v = (k < 300) ? W[n * 300 + k] : 0.f;
    packedW[idx] = (_Float16)v;
}

// ---------------------------------------------------------------------------
// One recurrence step: gates = emb_t @ W_ih^T + h @ W_hh^T + (b_ih+b_hh)
// Wave tile: 32 (M) x 48 (N). K = 320 (emb f16 gather) + 320 (h f16).
// ---------------------------------------------------------------------------
__global__ void __launch_bounds__(256)
lstm_step_gemm(const int* cap, const _Float16* embF16, const _Float16* packedW,
               const _Float16* hF16, const float* biasSum, float* gates, int t) {
    const int MT  = BSZ / 32;   // 128 wave rows
    const int NWT = NT / 3;     // 25 wave cols
    int wid = (blockIdx.x * blockDim.x + threadIdx.x) >> 5;
    if (wid >= MT * NWT) return;
    int lane = threadIdx.x & 31;
    int mt = wid % MT;
    int nw = wid / MT;
    int rowBase = mt * 32;
    int colBase = nw * 48;
    int lm16 = lane & 15;
    int c0   = (lane >> 4) * 8;

    // C init = broadcast bias (same n for all 8 rows held by a lane)
    v8f acc[2][3];
#pragma unroll
    for (int tj = 0; tj < 3; ++tj) {
        float bv = biasSum[colBase + tj * 16 + lm16];
#pragma unroll
        for (int ti = 0; ti < 2; ++ti)
#pragma unroll
            for (int r = 0; r < 8; ++r) acc[ti][tj][r] = bv;
    }

    // Embedding rows for this wave's two A tiles (f16 gather; table is in L2)
    int b0 = rowBase + lm16;
    int b1 = b0 + 16;
    const _Float16* er0 = embF16 + (size_t)cap[b0 * TLEN + t] * KPAD;
    const _Float16* er1 = embF16 + (size_t)cap[b1 * TLEN + t] * KPAD;

    // ---- K part 1: input projection over E (padded to 320) ----
#pragma unroll 2
    for (int kt = 0; kt < KT; ++kt) {
        int kb = kt * 32;
        v16h a0 = load_a_f16pad(er0, kb, c0);
        v16h a1 = load_a_f16pad(er1, kb, c0);
        const _Float16* tb = packedW + (size_t)(kt * NT + nw * 3) * TILE_HALFS + lane * 16;
        v16h bm0 = *(const v16h*)(tb);
        v16h bm1 = *(const v16h*)(tb + TILE_HALFS);
        v16h bm2 = *(const v16h*)(tb + 2 * TILE_HALFS);
        acc[0][0] = __builtin_amdgcn_wmma_f32_16x16x32_f16(false, a0, false, bm0, (short)0, acc[0][0], false, false);
        acc[0][1] = __builtin_amdgcn_wmma_f32_16x16x32_f16(false, a0, false, bm1, (short)0, acc[0][1], false, false);
        acc[0][2] = __builtin_amdgcn_wmma_f32_16x16x32_f16(false, a0, false, bm2, (short)0, acc[0][2], false, false);
        acc[1][0] = __builtin_amdgcn_wmma_f32_16x16x32_f16(false, a1, false, bm0, (short)0, acc[1][0], false, false);
        acc[1][1] = __builtin_amdgcn_wmma_f32_16x16x32_f16(false, a1, false, bm1, (short)0, acc[1][1], false, false);
        acc[1][2] = __builtin_amdgcn_wmma_f32_16x16x32_f16(false, a1, false, bm2, (short)0, acc[1][2], false, false);
    }

    // ---- K part 2: recurrent term over H (padded f16 rows, stride KPAD) ----
    const _Float16* hr0 = hF16 + (size_t)b0 * KPAD;
    const _Float16* hr1 = hF16 + (size_t)b1 * KPAD;
#pragma unroll 2
    for (int kt = 0; kt < KT; ++kt) {
        int kb = kt * 32;
        v16h a0 = load_a_f16pad(hr0, kb, c0);
        v16h a1 = load_a_f16pad(hr1, kb, c0);
        const _Float16* tb = packedW + (size_t)((KT + kt) * NT + nw * 3) * TILE_HALFS + lane * 16;
        v16h bm0 = *(const v16h*)(tb);
        v16h bm1 = *(const v16h*)(tb + TILE_HALFS);
        v16h bm2 = *(const v16h*)(tb + 2 * TILE_HALFS);
        acc[0][0] = __builtin_amdgcn_wmma_f32_16x16x32_f16(false, a0, false, bm0, (short)0, acc[0][0], false, false);
        acc[0][1] = __builtin_amdgcn_wmma_f32_16x16x32_f16(false, a0, false, bm1, (short)0, acc[0][1], false, false);
        acc[0][2] = __builtin_amdgcn_wmma_f32_16x16x32_f16(false, a0, false, bm2, (short)0, acc[0][2], false, false);
        acc[1][0] = __builtin_amdgcn_wmma_f32_16x16x32_f16(false, a1, false, bm0, (short)0, acc[1][0], false, false);
        acc[1][1] = __builtin_amdgcn_wmma_f32_16x16x32_f16(false, a1, false, bm1, (short)0, acc[1][1], false, false);
        acc[1][2] = __builtin_amdgcn_wmma_f32_16x16x32_f16(false, a1, false, bm2, (short)0, acc[1][2], false, false);
    }

    // Store per C/D layout: lane holds n = colBase+tj*16+L%16, m = r + 8*(L/16)
    int mBase = rowBase + (lane >> 4) * 8;
#pragma unroll
    for (int ti = 0; ti < 2; ++ti)
#pragma unroll
        for (int tj = 0; tj < 3; ++tj)
#pragma unroll
            for (int r = 0; r < 8; ++r) {
                int m = mBase + ti * 16 + r;
                int n = colBase + tj * 16 + lm16;
                gates[(size_t)m * G4 + n] = acc[ti][tj][r];
            }
}

// ---------------------------------------------------------------------------
// LSTM cell (torch gate order i,f,g,o); also captures h_last at t==len-1
// ---------------------------------------------------------------------------
__global__ void lstm_cell(const float* gates, const int* cap_len,
                          float* cbuf, _Float16* hF16, float* h_last, int t) {
    int idx = blockIdx.x * blockDim.x + threadIdx.x;
    if (idx >= BSZ * HDIM) return;
    int b = idx / HDIM;
    int k = idx - b * HDIM;
    const float* g = gates + (size_t)b * G4;
    float gi = 1.f / (1.f + expf(-g[k]));
    float gf = 1.f / (1.f + expf(-g[k + 300]));
    float gg = tanhf(g[k + 600]);
    float go = 1.f / (1.f + expf(-g[k + 900]));
    float cn = gf * cbuf[idx] + gi * gg;
    cbuf[idx] = cn;
    float h = go * tanhf(cn);
    hF16[(size_t)b * KPAD + k] = (_Float16)h;
    if (cap_len[b] - 1 == t) h_last[idx] = h;
}

// ---------------------------------------------------------------------------
// out[b, cls] = h_last[b] . Wn[cls] + cls_b[cls]    (NCLS = 2, tiny)
// ---------------------------------------------------------------------------
__global__ void classifier(const float* h_last, const float* Wn,
                           const float* cls_b, float* out) {
    int idx = blockIdx.x * blockDim.x + threadIdx.x;
    if (idx >= BSZ * 2) return;
    int b = idx >> 1, cls = idx & 1;
    const float* hv = h_last + (size_t)b * HDIM;
    const float* w  = Wn + cls * HDIM;
    float s = cls_b[cls];
    for (int k = 0; k < HDIM; ++k) s += hv[k] * w[k];
    out[idx] = s;
}

// ---------------------------------------------------------------------------
extern "C" void kernel_launch(void* const* d_in, const int* in_sizes, int n_in,
                              void* d_out, int out_size, void* d_ws, size_t ws_size,
                              hipStream_t stream) {
    const int*   cap     = (const int*)d_in[0];
    const int*   cap_len = (const int*)d_in[1];
    const float* embed_w = (const float*)d_in[2];
    const float* W_ih    = (const float*)d_in[3];
    const float* W_hh    = (const float*)d_in[4];
    const float* b_ih    = (const float*)d_in[5];
    const float* b_hh    = (const float*)d_in[6];
    const float* cls_v   = (const float*)d_in[7];
    const float* cls_g   = (const float*)d_in[8];
    const float* cls_b   = (const float*)d_in[9];
    float* out = (float*)d_out;

    // Workspace carve-up (aligned to 256B); total ~40 MB
    char*  base = (char*)d_ws;
    size_t off  = 0;
    auto carve = [&](size_t bytes) -> void* {
        void* r = base + off;
        off = (off + bytes + 255) & ~(size_t)255;
        return r;
    };
    _Float16* packedW = (_Float16*)carve((size_t)2 * KT * NT * TILE_HALFS * sizeof(_Float16));
    _Float16* embF16  = (_Float16*)carve((size_t)VOCAB * KPAD * sizeof(_Float16));
    _Float16* hF16    = (_Float16*)carve((size_t)BSZ * KPAD * sizeof(_Float16));
    float*    cbuf    = (float*)carve((size_t)BSZ * HDIM * sizeof(float));
    float*    h_last  = (float*)carve((size_t)BSZ * HDIM * sizeof(float));
    float*    gates   = (float*)carve((size_t)BSZ * G4 * sizeof(float));
    float*    biasSum = (float*)carve((size_t)G4 * sizeof(float));
    float*    Wn      = (float*)carve((size_t)2 * HDIM * sizeof(float));
    (void)ws_size; (void)in_sizes; (void)n_in; (void)out_size;

    // 1) init h=0 (incl. pad cols), c=0
    {
        int n = BSZ * KPAD;
        init_state<<<(n + 255) / 256, 256, 0, stream>>>(hF16, cbuf);
    }
    // 2) embedding table -> padded f16
    {
        int n = VOCAB * KPAD;
        cvt_embed<<<(n + 255) / 256, 256, 0, stream>>>(embed_w, embF16);
    }
    // 3) bias sum + weight-normed classifier weights
    prep_cls_bias<<<1, 256, 0, stream>>>(b_ih, b_hh, cls_v, cls_g, biasSum, Wn);
    // 4) pack W_ih / W_hh into WMMA B-fragment tiles
    {
        int n = 2 * KT * NT * TILE_HALFS;
        pack_weights<<<(n + 255) / 256, 256, 0, stream>>>(W_ih, W_hh, packedW);
    }
    // 5) 32 recurrence steps
    const int gemmWaves   = (BSZ / 32) * (NT / 3);          // 3200
    const int gemmBlocks  = (gemmWaves * 32 + 255) / 256;   // 400
    const int cellBlocks  = (BSZ * HDIM + 255) / 256;       // 4800
    for (int t = 0; t < TLEN; ++t) {
        lstm_step_gemm<<<gemmBlocks, 256, 0, stream>>>(cap, embF16, packedW,
                                                       hF16, biasSum, gates, t);
        lstm_cell<<<cellBlocks, 256, 0, stream>>>(gates, cap_len, cbuf, hF16, h_last, t);
    }
    // 6) classifier
    classifier<<<(BSZ * 2 + 255) / 256, 256, 0, stream>>>(h_last, Wn, cls_b, out);
}
